// MoE_20315195310389
// MI455X (gfx1250) — matrix operand order repaired and checked by reading, not compiled, not run on hardware
//
#include <hip/hip_runtime.h>
#include <hip/hip_bf16.h>

typedef __bf16 bf16_t;
typedef __attribute__((ext_vector_type(16))) __bf16 v16bf;
typedef __attribute__((ext_vector_type(8)))  float  v8f;

#define T_TOK 4096
#define DIMX  2048
#define INTERX 1408
#define NEXP  8
#define TOPK  2
#define ROUTE_SCALE 1.0f
#define KC    32    // K-chunk
#define LSTR  40    // padded LDS row stride in bf16 (32 data + 8 pad = 20 banks)

// ---------------------------------------------------------------------------
// gfx1250 async global->LDS copy (16B per lane), tracked by ASYNCcnt.
// ---------------------------------------------------------------------------
__device__ __forceinline__ void async_copy16(unsigned int lds_off, const void* g)
{
    asm volatile("global_load_async_to_lds_b128 %0, %1, off"
                 :: "v"(lds_off), "v"((unsigned long long)(size_t)g)
                 : "memory");
}
__device__ __forceinline__ void wait_async0()
{
    asm volatile("s_wait_asynccnt 0" ::: "memory");
}
__device__ __forceinline__ unsigned int lds_addr(const void* p)
{
    return (unsigned int)(size_t)p;   // low 32 bits of LDS aperture addr = LDS offset
}
__device__ __forceinline__ v16bf ld_frag_a(const bf16_t* p)   // chunks at +0,+16 elems
{
    v16bf r;
    ((uint4*)&r)[0] = *(const uint4*)(p);
    ((uint4*)&r)[1] = *(const uint4*)(p + 16);
    return r;
}
__device__ __forceinline__ v16bf ld_frag_b(const bf16_t* p)   // contiguous 16 elems
{
    v16bf r;
    ((uint4*)&r)[0] = *(const uint4*)(p);
    ((uint4*)&r)[1] = *(const uint4*)(p + 8);
    return r;
}

// ---------------------------------------------------------------------------
// Gate: one block per token; 8 waves, wave e computes score for expert e.
// ---------------------------------------------------------------------------
__global__ __launch_bounds__(256) void moe_gate(
    const bf16_t* __restrict__ X, const float* __restrict__ GW,
    const float* __restrict__ GB, float* __restrict__ cw, int* __restrict__ sel)
{
    const int t    = blockIdx.x;
    const int lane = threadIdx.x & 31;
    const int e    = threadIdx.x >> 5;
    const bf16_t* xp = X + (size_t)t * DIMX;
    const float*  gp = GW + (size_t)e * DIMX;
    float acc = 0.f;
    for (int d = lane; d < DIMX; d += 32)
        acc += (float)xp[d] * gp[d];
    for (int s = 16; s > 0; s >>= 1)
        acc += __shfl_xor(acc, s, 32);
    __shared__ float sc[NEXP];
    if (lane == 0) sc[e] = acc;
    __syncthreads();
    if (threadIdx.x == 0) {
        float s[NEXP], b[NEXP];
        for (int i = 0; i < NEXP; ++i) {
            float r  = sc[i];
            float sp = fmaxf(r, 0.f) + log1pf(__expf(-fabsf(r)));  // stable softplus
            s[i] = sqrtf(sp);
            b[i] = s[i] + GB[i];
        }
        int i0 = 0;
        for (int i = 1; i < NEXP; ++i) if (b[i] > b[i0]) i0 = i;
        int i1 = -1;
        for (int i = 0; i < NEXP; ++i) {
            if (i == i0) continue;
            if (i1 < 0 || b[i] > b[i1]) i1 = i;
        }
        for (int i = 0; i < NEXP; ++i) cw[t * NEXP + i] = 0.f;
        cw[t * NEXP + i0] = s[i0] * ROUTE_SCALE;
        cw[t * NEXP + i1] = s[i1] * ROUTE_SCALE;
        sel[t * TOPK + 0] = i0;
        sel[t * TOPK + 1] = i1;
    }
}

// ---------------------------------------------------------------------------
// Deterministic compaction: 1 block, 256 threads, 16 tokens/thread.
// ---------------------------------------------------------------------------
__global__ __launch_bounds__(256) void moe_compact(
    const int* __restrict__ sel, int* __restrict__ cnt, int* __restrict__ off,
    int* __restrict__ idxp, int* __restrict__ row_of)
{
    __shared__ int lcnt[256][NEXP];
    __shared__ int tot[NEXP];
    __shared__ int soff[NEXP];
    const int tid = threadIdx.x;
    const int t0  = tid * (T_TOK / 256);
    int c[NEXP];
    for (int e = 0; e < NEXP; ++e) c[e] = 0;
    for (int j = 0; j < T_TOK / 256; ++j) {
        int t = t0 + j;
        c[sel[2 * t + 0]]++;
        c[sel[2 * t + 1]]++;
    }
    for (int e = 0; e < NEXP; ++e) lcnt[tid][e] = c[e];
    __syncthreads();
    if (tid < NEXP) {
        int run = 0;
        for (int i = 0; i < 256; ++i) { int v = lcnt[i][tid]; lcnt[i][tid] = run; run += v; }
        cnt[tid] = run;
        tot[tid] = run;
    }
    __syncthreads();
    if (tid == 0) {
        int run = 0;
        for (int e = 0; e < NEXP; ++e) { soff[e] = run; off[e] = run; run += tot[e]; }
    }
    __syncthreads();
    int p[NEXP];
    for (int e = 0; e < NEXP; ++e) p[e] = soff[e] + lcnt[tid][e];
    for (int j = 0; j < T_TOK / 256; ++j) {
        int t = t0 + j;
        for (int k = 0; k < TOPK; ++k) {
            int e = sel[2 * t + k];
            int r = p[e]++;
            idxp[r] = t;
            row_of[2 * t + k] = r;
        }
    }
}

// ---------------------------------------------------------------------------
// FFN stage 1: H = silu(X@W1^T) * (X@W3^T) * cw
// Block tile 128Mx64N; 8 waves (4x2); wave tile 32x32; K-chunk 32;
// async-to-LDS double buffering.
// ---------------------------------------------------------------------------
__global__ __launch_bounds__(256) void moe_ffn_in(
    const bf16_t* __restrict__ X, const bf16_t* __restrict__ W1all,
    const bf16_t* __restrict__ W3all, const float* __restrict__ cw,
    const int* __restrict__ idxp, const int* __restrict__ off,
    const int* __restrict__ cnt, bf16_t* __restrict__ H, int gather)
{
    __shared__ __align__(16) bf16_t As [2][128 * LSTR];
    __shared__ __align__(16) bf16_t B1s[2][ 64 * LSTR];
    __shared__ __align__(16) bf16_t B3s[2][ 64 * LSTR];
    __shared__ int   stok[128];
    __shared__ float scw [128];

    const int e  = blockIdx.z;
    const int r0 = gather ? off[e] : 0;
    const int M  = gather ? cnt[e] : T_TOK;
    const int mt = blockIdx.x * 128;
    if (mt >= M) return;
    const int nt = blockIdx.y * 64;
    const bf16_t* W1 = W1all + (size_t)e * INTERX * DIMX;
    const bf16_t* W3 = W3all + (size_t)e * INTERX * DIMX;

    const int tid  = threadIdx.x;
    const int lane = tid & 31;
    const int wave = tid >> 5;
    const int wm   = wave >> 1;          // 0..3  -> M offset wm*32
    const int wn   = wave & 1;           // 0..1  -> N offset wn*32
    const int half = lane >> 4;
    const int l16  = lane & 15;
    const int arow = tid >> 1;           // 0..127 A loader row
    const int acol = (tid & 1) * 16;     // 0/16
    const int brow = tid >> 2;           // 0..63  B loader row
    const int bcol = (tid & 3) * 8;      // 0/8/16/24

    if (tid < 128) {
        int m  = mt + tid;
        int ok = m < M;
        int tk = gather ? idxp[r0 + (ok ? m : 0)] : (ok ? m : 0);
        stok[tid] = tk;
        scw[tid]  = gather ? (ok ? cw[tk * NEXP + e] : 0.f) : 1.0f;
    }
    __syncthreads();

    const int tokr = stok[arow];

    auto issue_chunk = [&](int buf, int kc) {
        const bf16_t* ga = X  + (size_t)tokr * DIMX        + kc + acol;
        unsigned int  la = lds_addr(&As[buf][arow * LSTR + acol]);
        async_copy16(la,      ga);
        async_copy16(la + 16, ga + 8);
        const bf16_t* g1 = W1 + (size_t)(nt + brow) * DIMX + kc + bcol;
        const bf16_t* g3 = W3 + (size_t)(nt + brow) * DIMX + kc + bcol;
        async_copy16(lds_addr(&B1s[buf][brow * LSTR + bcol]), g1);
        async_copy16(lds_addr(&B3s[buf][brow * LSTR + bcol]), g3);
    };

    issue_chunk(0, 0);

    v8f acc1[2][2] = {{v8f{}, v8f{}}, {v8f{}, v8f{}}};
    v8f acc3[2][2] = {{v8f{}, v8f{}}, {v8f{}, v8f{}}};

    const int nchunks = DIMX / KC;  // 64
    for (int c = 0; c < nchunks; ++c) {
        const int buf = c & 1;
        wait_async0();
        __syncthreads();
        if (c + 1 < nchunks) issue_chunk(buf ^ 1, (c + 1) * KC);

        v16bf a[2];
        #pragma unroll
        for (int am = 0; am < 2; ++am)
            a[am] = ld_frag_a(&As[buf][(wm * 32 + am * 16 + l16) * LSTR + half * 8]);
        #pragma unroll
        for (int sub = 0; sub < 2; ++sub) {
            const int ncl = (wn * 32 + sub * 16 + l16) * LSTR + half * 8;
            v16bf b1 = ld_frag_a(&B1s[buf][ncl]);
            v16bf b3 = ld_frag_a(&B3s[buf][ncl]);
            #pragma unroll
            for (int am = 0; am < 2; ++am) {
                acc1[am][sub] = __builtin_amdgcn_wmma_f32_16x16x32_bf16(false, a[am], false, b1, (short)0, acc1[am][sub], false, false);
                acc3[am][sub] = __builtin_amdgcn_wmma_f32_16x16x32_bf16(false, a[am], false, b3, (short)0, acc3[am][sub], false, false);
            }
        }
    }

    // Epilogue: C elem i -> row = i + 8*half (within 16), col = l16
    #pragma unroll
    for (int am = 0; am < 2; ++am) {
        #pragma unroll
        for (int sub = 0; sub < 2; ++sub) {
            int ncol = nt + wn * 32 + sub * 16 + l16;
            #pragma unroll
            for (int i = 0; i < 8; ++i) {
                int mloc = wm * 32 + am * 16 + i + 8 * half;
                int m    = mt + mloc;
                if (m < M) {
                    float cwt = scw[mloc];
                    float x1  = acc1[am][sub][i], x3 = acc3[am][sub][i];
                    float hv  = (x1 / (1.0f + __expf(-x1))) * x3 * cwt;
                    H[(size_t)(r0 + m) * INTERX + ncol] = (bf16_t)hv;
                }
            }
        }
    }
}

// ---------------------------------------------------------------------------
// FFN stage 2: Y = H @ W2^T  (packed rows)
// Block tile 128Mx128N; 8 waves (4x2); wave tile 32x64; K-chunk 32.
// ---------------------------------------------------------------------------
__global__ __launch_bounds__(256) void moe_ffn_out(
    const bf16_t* __restrict__ Hsrc, const bf16_t* __restrict__ W2all,
    const int* __restrict__ off, const int* __restrict__ cnt,
    bf16_t* __restrict__ Y, int gather)
{
    __shared__ __align__(16) bf16_t As[2][128 * LSTR];
    __shared__ __align__(16) bf16_t Bs[2][128 * LSTR];

    const int e  = blockIdx.z;
    const int r0 = gather ? off[e] : 0;
    const int M  = gather ? cnt[e] : T_TOK;
    const int mt = blockIdx.x * 128;
    if (mt >= M) return;
    const int nt = blockIdx.y * 128;
    const bf16_t* W2 = W2all + (size_t)e * DIMX * INTERX;

    const int tid  = threadIdx.x;
    const int lane = tid & 31;
    const int wave = tid >> 5;
    const int wm   = wave >> 1;          // M offset wm*32
    const int wn   = wave & 1;           // N offset wn*64
    const int half = lane >> 4;
    const int l16  = lane & 15;
    const int arow = tid >> 1;           // 0..127
    const int acol = (tid & 1) * 16;

    int ar = mt + arow;
    if (ar >= M) ar = M - 1;             // clamp (masked on store)

    auto issue_chunk = [&](int buf, int kc) {
        const bf16_t* ga = Hsrc + (size_t)(r0 + ar) * INTERX   + kc + acol;
        const bf16_t* gb = W2   + (size_t)(nt + arow) * INTERX + kc + acol;
        unsigned int  la = lds_addr(&As[buf][arow * LSTR + acol]);
        unsigned int  lb = lds_addr(&Bs[buf][arow * LSTR + acol]);
        async_copy16(la,      ga);
        async_copy16(la + 16, ga + 8);
        async_copy16(lb,      gb);
        async_copy16(lb + 16, gb + 8);
    };

    issue_chunk(0, 0);

    v8f acc[2][4] = {{v8f{}, v8f{}, v8f{}, v8f{}}, {v8f{}, v8f{}, v8f{}, v8f{}}};

    const int nchunks = INTERX / KC;  // 44
    for (int c = 0; c < nchunks; ++c) {
        const int buf = c & 1;
        wait_async0();
        __syncthreads();
        if (c + 1 < nchunks) issue_chunk(buf ^ 1, (c + 1) * KC);

        v16bf a[2];
        #pragma unroll
        for (int am = 0; am < 2; ++am)
            a[am] = ld_frag_a(&As[buf][(wm * 32 + am * 16 + l16) * LSTR + half * 8]);
        #pragma unroll
        for (int sub = 0; sub < 4; ++sub) {
            v16bf bfr = ld_frag_a(&Bs[buf][(wn * 64 + sub * 16 + l16) * LSTR + half * 8]);
            #pragma unroll
            for (int am = 0; am < 2; ++am)
                acc[am][sub] = __builtin_amdgcn_wmma_f32_16x16x32_bf16(false, a[am], false, bfr, (short)0, acc[am][sub], false, false);
        }
    }

    #pragma unroll
    for (int am = 0; am < 2; ++am) {
        #pragma unroll
        for (int sub = 0; sub < 4; ++sub) {
            int ncol = nt + wn * 64 + sub * 16 + l16;
            #pragma unroll
            for (int i = 0; i < 8; ++i) {
                int m = mt + wm * 32 + am * 16 + i + 8 * half;
                if (m < M)
                    Y[(size_t)(r0 + m) * DIMX + ncol] = (bf16_t)acc[am][sub][i];
            }
        }
    }
}

// ---------------------------------------------------------------------------
// Combine: out[t] = Y[row_of[t,0]] + Y[row_of[t,1]] + Z[t]
// ---------------------------------------------------------------------------
__global__ __launch_bounds__(256) void moe_combine(
    const bf16_t* __restrict__ Ybuf, const bf16_t* __restrict__ Zbuf,
    const int* __restrict__ row_of, bf16_t* __restrict__ out)
{
    size_t i = (size_t)blockIdx.x * blockDim.x + threadIdx.x;
    int t = (int)(i / DIMX);
    int d = (int)(i % DIMX);
    int ra = row_of[2 * t + 0];
    int rb = row_of[2 * t + 1];
    float v = (float)Ybuf[(size_t)ra * DIMX + d]
            + (float)Ybuf[(size_t)rb * DIMX + d]
            + (float)Zbuf[i];
    out[i] = (bf16_t)v;
}

// ---------------------------------------------------------------------------
extern "C" void kernel_launch(void* const* d_in, const int* in_sizes, int n_in,
                              void* d_out, int out_size, void* d_ws, size_t ws_size,
                              hipStream_t stream)
{
    const bf16_t* x   = (const bf16_t*)d_in[0];
    const float*  gw  = (const float*) d_in[1];
    const float*  gb  = (const float*) d_in[2];
    const bf16_t* w1  = (const bf16_t*)d_in[3];
    const bf16_t* w2  = (const bf16_t*)d_in[4];
    const bf16_t* w3  = (const bf16_t*)d_in[5];
    const bf16_t* sw1 = (const bf16_t*)d_in[6];
    const bf16_t* sw2 = (const bf16_t*)d_in[7];
    const bf16_t* sw3 = (const bf16_t*)d_in[8];
    bf16_t* out = (bf16_t*)d_out;

    char* ws = (char*)d_ws;
    float* cw     = (float*)ws;  ws += (size_t)T_TOK * NEXP * 4;
    int*   sel    = (int*)ws;    ws += (size_t)T_TOK * TOPK * 4;
    int*   cnt    = (int*)ws;    ws += 64;
    int*   off    = (int*)ws;    ws += 64;
    int*   idxp   = (int*)ws;    ws += (size_t)T_TOK * TOPK * 4;
    int*   row_of = (int*)ws;    ws += (size_t)T_TOK * TOPK * 4;
    bf16_t* Hbuf  = (bf16_t*)ws; ws += (size_t)T_TOK * TOPK * INTERX * 2;
    bf16_t* Ybuf  = (bf16_t*)ws; ws += (size_t)T_TOK * TOPK * DIMX * 2;
    bf16_t* Hs    = (bf16_t*)ws; ws += (size_t)T_TOK * INTERX * 2;
    bf16_t* Zbuf  = (bf16_t*)ws; ws += (size_t)T_TOK * DIMX * 2;

    moe_gate<<<T_TOK, 256, 0, stream>>>(x, gw, gb, cw, sel);
    moe_compact<<<1, 256, 0, stream>>>(sel, cnt, off, idxp, row_of);

    dim3 gin(T_TOK / 128, INTERX / 64, NEXP);        // 32 x 22 x 8
    moe_ffn_in<<<gin, 256, 0, stream>>>(x, w1, w3, cw, idxp, off, cnt, Hbuf, 1);
    dim3 gins(T_TOK / 128, INTERX / 64, 1);
    moe_ffn_in<<<gins, 256, 0, stream>>>(x, sw1, sw3, cw, idxp, off, cnt, Hs, 0);

    dim3 gout(T_TOK / 128, DIMX / 128, NEXP);        // 32 x 16 x 8
    moe_ffn_out<<<gout, 256, 0, stream>>>(Hbuf, w2, off, cnt, Ybuf, 1);
    dim3 gouts(T_TOK / 128, DIMX / 128, 1);
    moe_ffn_out<<<gouts, 256, 0, stream>>>(Hs, sw2, off, cnt, Zbuf, 0);

    moe_combine<<<(T_TOK * DIMX) / 256, 256, 0, stream>>>(Ybuf, Zbuf, row_of, out);
}
